// Controller_29076928594279
// MI455X (gfx1250) — matrix operand order repaired and checked by reading, not compiled, not run on hardware
//
#include <hip/hip_runtime.h>
#include <hip/hip_bf16.h>
#include <math.h>

typedef __attribute__((ext_vector_type(16))) _Float16 v16h;
typedef __attribute__((ext_vector_type(8)))  float    v8f;

#define THREADS 256
#define NWAVES  8

// ---------------- shared memory layout ----------------
// f16 region (units: halves)
#define FRAG_WCAT 0        // 256x128 fused [W_ih | W_hh] A-fragments : 32768 halves
#define FRAG_A1   32768    // w_attn1 64x64                           : 4096
#define FRAG_A2   36864    // w_attn2 64x64                           : 4096
#define FRAG_WS   40960    // w_soft padded 16x64                     : 1024
#define VIN16     41984    // lstm input vector [x;h] as f16          : 128
#define HV16      42112    // h as f16 for attn/soft matvecs          : 64
#define HALF_END  42176
// f32 region (units: floats, offset from float base)
#define G_      0          // 256 : fused gate pre-activations
#define HQ_     256        // 64  : h @ w_attn2^T
#define WSOUT_  320        // 16  : h @ w_soft^T (padded)
#define X_      336        // 64  : current LSTM input
#define H_      400        // 64
#define C_      464        // 64
#define ANCH_   528        // 7*64 anchors
#define ANW1_   976        // 7*64 anchors @ w_attn1^T
#define LOGITS_ 1424       // 8
#define BIH_    1432       // 256
#define BHH_    1688       // 256
#define EMB_    1944       // 6*64
#define BS_     2328       // 5 (pad 8)
#define BSN_    2336       // 5 (pad 8)
#define VATT_   2344       // 64
#define SCAL_   2408       // [0]=log_prob [1]=entropy
#define FLOAT_END 2416
// int region
#define IARC    0          // 40 arc entries
#define IRNG    40
#define ICHOSEN 41
#define ISEED   42
#define INT_END 48

#define SMEM_BYTES (HALF_END*2 + FLOAT_END*4 + INT_END*4)

// ---- A-fragment index -> (row, col) of source matrix (16-bit A, contiguous k = 16*(lane/16) + j) ----
__device__ __forceinline__ void frag_rc(int f, int Ktiles, int& row, int& col) {
  int tile   = f >> 9;        // 512 halves per 16x32 tile
  int within = f & 511;
  int lane   = within >> 4;   // 0..31
  int j      = within & 15;   // half index within lane's v16h
  int mt = tile / Ktiles, kt = tile % Ktiles;
  int hg = lane >> 4, m = lane & 15;
  int p = j >> 1, half = j & 1;
  int k = 16 * hg + j;
  row = mt * 16 + m;
  col = kt * 32 + k;
}

__device__ __forceinline__ float sigmoidf_(float x) { return 1.0f / (1.0f + expf(-x)); }

__device__ __forceinline__ float u01_(unsigned long long s) {
  s += 0x9E3779B97F4A7C15ull;
  s = (s ^ (s >> 30)) * 0xBF58476D1CE4E5B9ull;
  s = (s ^ (s >> 27)) * 0x94D049BB133111EBull;
  s ^= s >> 31;
  unsigned int r = (unsigned int)(s >> 40);               // 24 random bits
  return ((float)r + 0.5f) * (1.0f / 16777216.0f);
}

// out[R] = Wfrag[R,K] @ vin[K]; vector broadcast across all 16 B columns.
// Wave w handles M-tiles w, w+8, ...; K accumulated through the WMMA C operand.
__device__ __forceinline__ void wmma_matvec(const _Float16* sh, int fragBase,
                                            int Mtiles, int Ktiles,
                                            int vinBase, float* out, int tid) {
  int wave = tid >> 5, lane = tid & 31;
  int hg = lane >> 4;                 // 0: K 0..15 of chunk, 1: K 16..31 (B layout)
  for (int mt = wave; mt < Mtiles; mt += NWAVES) {
    v8f acc = {};
    for (int kt = 0; kt < Ktiles; ++kt) {
      v16h a = *(const v16h*)(sh + fragBase + (mt * Ktiles + kt) * 512 + lane * 16);
      v16h b = *(const v16h*)(sh + vinBase + kt * 32 + hg * 16);
      acc = __builtin_amdgcn_wmma_f32_16x16x32_f16(false, a, false, b,
                                                   (short)0, acc, false, false);
    }
    if ((lane & 15) == 0) {           // column N=0 carries rows mt*16+hg*8 .. +7
      int rbase = mt * 16 + hg * 8;
#pragma unroll
      for (int r = 0; r < 8; ++r) out[rbase + r] = acc[r];
    }
  }
}

// One LSTM cell step: g = Wcat @ [x;h] (+biases), then gate math on 64 threads.
__device__ void lstm_step(_Float16* sh, float* sf, int tid) {
  if (tid < 128) {
    float v = (tid < 64) ? sf[X_ + tid] : sf[H_ + tid - 64];
    sh[VIN16 + tid] = (_Float16)v;
  }
  __syncthreads();
  wmma_matvec(sh, FRAG_WCAT, 16, 4, VIN16, sf + G_, tid);
  __syncthreads();
  if (tid < 64) {
    float gi = sf[G_ + tid]       + sf[BIH_ + tid]       + sf[BHH_ + tid];
    float gf = sf[G_ + 64 + tid]  + sf[BIH_ + 64 + tid]  + sf[BHH_ + 64 + tid];
    float gg = sf[G_ + 128 + tid] + sf[BIH_ + 128 + tid] + sf[BHH_ + 128 + tid];
    float go = sf[G_ + 192 + tid] + sf[BIH_ + 192 + tid] + sf[BHH_ + 192 + tid];
    float c2 = sigmoidf_(gf) * sf[C_ + tid] + sigmoidf_(gi) * tanhf(gg);
    float h2 = sigmoidf_(go) * tanhf(c2);
    sf[C_ + tid] = c2;
    sf[H_ + tid] = h2;
  }
  __syncthreads();
}

// out[..] = Wfrag @ h   (h converted to f16 first)
__device__ void h_matvec(_Float16* sh, float* sf, int fragBase, int Mtiles,
                         float* out, int tid) {
  if (tid < 64) sh[HV16 + tid] = (_Float16)sf[H_ + tid];
  __syncthreads();
  wmma_matvec(sh, fragBase, Mtiles, 2, HV16, out, tid);
  __syncthreads();
}

// Gumbel-max categorical over sf[LOGITS_ .. +nc), thread 0 only.
__device__ void sample_cat(float* sf, int* si, int nc) {
  int cnt = si[IRNG]++;
  unsigned long long base =
      ((unsigned long long)(unsigned int)si[ISEED] << 32) ^
      ((unsigned long long)(unsigned int)cnt * 0x00010003ull);
  int best = 0; float bestv = -3.0e38f;
  for (int j = 0; j < nc; ++j) {
    float u = u01_(base + (unsigned long long)j * 0x9E37ull);
    float g = -logf(-logf(u));
    float v = sf[LOGITS_ + j] + g;
    if (v > bestv) { bestv = v; best = j; }
  }
  float m = sf[LOGITS_];
  for (int j = 1; j < nc; ++j) m = fmaxf(m, sf[LOGITS_ + j]);
  float se = 0.0f;
  for (int j = 0; j < nc; ++j) se += expf(sf[LOGITS_ + j] - m);
  float lse = m + logf(se);
  sf[SCAL_ + 0] += -(sf[LOGITS_ + best] - lse);    // log_prob
  float ent = 0.0f;
  for (int j = 0; j < nc; ++j) {
    float lp = sf[LOGITS_ + j] - lse;
    ent -= expf(lp) * lp;
  }
  sf[SCAL_ + 1] += ent;                            // entropy
  si[ICHOSEN] = best;
}

__device__ void sampler(_Float16* sh, float* sf, int* si, int tid,
                        int use_bias, int arcBase) {
  if (tid < 64) sf[X_ + tid] = sf[EMB_ + tid];     // inputs = embed[0]
  __syncthreads();
  // warmup: 2 zero anchors, but anchors_w1 = prev_h @ w_attn1^T
  for (int w = 0; w < 2; ++w) {
    lstm_step(sh, sf, tid);
    if (tid < 64) sf[ANCH_ + w * 64 + tid] = 0.0f;
    h_matvec(sh, sf, FRAG_A1, 4, sf + ANW1_ + w * 64, tid);
  }
  for (int layer = 2; layer < 7; ++layer) {
    // ---- 2 index samples via tiny attention ----
    for (int s = 0; s < 2; ++s) {
      lstm_step(sh, sf, tid);
      h_matvec(sh, sf, FRAG_A2, 4, sf + HQ_, tid);
      if (tid < layer) {
        float acc = 0.0f;
        for (int d = 0; d < 64; ++d)
          acc += sf[VATT_ + d] * tanhf(sf[ANW1_ + tid * 64 + d] + sf[HQ_ + d]);
        sf[LOGITS_ + tid] = acc / 5.0f + 1.1f * tanhf(acc);
      }
      __syncthreads();
      if (tid == 0) sample_cat(sf, si, layer);
      __syncthreads();
      int index = si[ICHOSEN];
      if (tid < 64) sf[X_ + tid] = sf[ANCH_ + index * 64 + tid];
      if (tid == 0) si[IARC + arcBase + (layer - 2) * 4 + 2 * s] = index;
      __syncthreads();
    }
    // ---- 2 op samples ----
    for (int s = 0; s < 2; ++s) {
      lstm_step(sh, sf, tid);
      h_matvec(sh, sf, FRAG_WS, 1, sf + WSOUT_, tid);
      if (tid == 0) {
        for (int j = 0; j < 5; ++j) {
          float l = sf[WSOUT_ + j] + sf[BS_ + j];
          l = (1.1f / 2.5f) * tanhf(l / 5.0f);
          if (use_bias) l += sf[BSN_ + j];
          sf[LOGITS_ + j] = l;
        }
        sample_cat(sf, si, 5);
      }
      __syncthreads();
      int op = si[ICHOSEN];
      if (tid < 64) sf[X_ + tid] = sf[EMB_ + (op + 1) * 64 + tid];
      if (tid == 0) si[IARC + arcBase + (layer - 2) * 4 + 2 * s + 1] = op;
      __syncthreads();
    }
    // ---- anchor step ----
    lstm_step(sh, sf, tid);
    if (tid < 64) sf[ANCH_ + layer * 64 + tid] = sf[H_ + tid];
    h_matvec(sh, sf, FRAG_A1, 4, sf + ANW1_ + layer * 64, tid);
    if (tid < 64) sf[X_ + tid] = sf[EMB_ + tid];
    __syncthreads();
  }
}

__global__ void __launch_bounds__(THREADS, 1)
enas_controller_kernel(const float* embed, const float* w_ih, const float* w_hh,
                       const float* b_ih, const float* b_hh, const float* w_soft,
                       const float* b_soft, const float* b_soft_nl,
                       const float* w_attn1, const float* w_attn2,
                       const float* v_attn, const int* seedp, float* out) {
  extern __shared__ char smem_raw[];
  _Float16* sh = (_Float16*)smem_raw;
  float*    sf = (float*)(smem_raw + HALF_END * 2);
  int*      si = (int*)(smem_raw + HALF_END * 2 + FLOAT_END * 4);
  int tid = threadIdx.x;

  // ---- one-time init: constants + scalars ----
  if (tid == 0) {
    sf[SCAL_ + 0] = 0.0f; sf[SCAL_ + 1] = 0.0f;
    si[IRNG] = 0; si[ISEED] = seedp[0];
  }
  if (tid < 64) { sf[H_ + tid] = 0.0f; sf[C_ + tid] = 0.0f; }
  for (int i = tid; i < 256; i += THREADS) { sf[BIH_ + i] = b_ih[i]; sf[BHH_ + i] = b_hh[i]; }
  for (int i = tid; i < 384; i += THREADS) sf[EMB_ + i] = embed[i];
  if (tid < 5) { sf[BS_ + tid] = b_soft[tid]; sf[BSN_ + tid] = b_soft_nl[tid]; }
  if (tid < 64) sf[VATT_ + tid] = v_attn[tid];

  // ---- pre-pack weights as f16 A-fragments in LDS ----
  int row, col;
  for (int f = tid; f < 256 * 128; f += THREADS) {         // Wcat = [W_ih | W_hh]
    frag_rc(f, 4, row, col);
    float v = (col < 64) ? w_ih[row * 64 + col] : w_hh[row * 64 + (col - 64)];
    sh[FRAG_WCAT + f] = (_Float16)v;
  }
  for (int f = tid; f < 64 * 64; f += THREADS) {           // w_attn1
    frag_rc(f, 2, row, col);
    sh[FRAG_A1 + f] = (_Float16)w_attn1[row * 64 + col];
  }
  for (int f = tid; f < 64 * 64; f += THREADS) {           // w_attn2
    frag_rc(f, 2, row, col);
    sh[FRAG_A2 + f] = (_Float16)w_attn2[row * 64 + col];
  }
  for (int f = tid; f < 16 * 64; f += THREADS) {           // w_soft padded to 16 rows
    frag_rc(f, 2, row, col);
    sh[FRAG_WS + f] = (_Float16)((row < 5) ? w_soft[row * 64 + col] : 0.0f);
  }
  __syncthreads();

  sampler(sh, sf, si, tid, /*use_bias=*/1, /*arcBase=*/0);   // keeps c,h for sampler 2
  sampler(sh, sf, si, tid, /*use_bias=*/0, /*arcBase=*/20);
  __syncthreads();

  if (tid < 40) out[tid] = (float)si[IARC + tid];            // arc1[20], arc2[20]
  if (tid == 0) { out[40] = sf[SCAL_ + 0]; out[41] = sf[SCAL_ + 1]; }
}

extern "C" void kernel_launch(void* const* d_in, const int* in_sizes, int n_in,
                              void* d_out, int out_size, void* d_ws, size_t ws_size,
                              hipStream_t stream) {
  (void)in_sizes; (void)n_in; (void)out_size; (void)d_ws; (void)ws_size;
  const float* embed     = (const float*)d_in[0];
  const float* w_ih      = (const float*)d_in[1];
  const float* w_hh      = (const float*)d_in[2];
  const float* b_ih      = (const float*)d_in[3];
  const float* b_hh      = (const float*)d_in[4];
  const float* w_soft    = (const float*)d_in[5];
  const float* b_soft    = (const float*)d_in[6];
  const float* b_soft_nl = (const float*)d_in[7];
  const float* w_attn1   = (const float*)d_in[8];
  const float* w_attn2   = (const float*)d_in[9];
  const float* v_attn    = (const float*)d_in[10];
  const int*   seedp     = (const int*)d_in[11];
  float* out = (float*)d_out;

  // Opt in to >64KB dynamic LDS (WGP supports 320KB); deterministic, capture-safe.
  (void)hipFuncSetAttribute((const void*)enas_controller_kernel,
                            hipFuncAttributeMaxDynamicSharedMemorySize, SMEM_BYTES);

  enas_controller_kernel<<<dim3(1), dim3(THREADS), SMEM_BYTES, stream>>>(
      embed, w_ih, w_hh, b_ih, b_hh, w_soft, b_soft, b_soft_nl,
      w_attn1, w_attn2, v_attn, seedp, out);
}
